// Mamba2Layer_30064771072945
// MI455X (gfx1250) — matrix operand order repaired and checked by reading, not compiled, not run on hardware
//
#include <hip/hip_runtime.h>

// ---------------------------------------------------------------------------
// Mamba2 forward for MI455X (gfx1250): bf16 WMMA everywhere, f32 accumulate.
// ---------------------------------------------------------------------------

#define D_MODEL   2048
#define D_STATE   128
#define D_CONV    4
#define D_INNER   4096
#define NHEADS    64
#define HEADDIM   64
#define CHUNK     256
#define NCHUNK    8
#define CONV_DIM  4352          // D_INNER + 2*D_STATE
#define D_IN_PROJ 8512          // 2*D_INNER + 2*D_STATE + NHEADS
#define BATCH     2
#define SEQLEN    2048
#define BL        (BATCH*SEQLEN)     // 4096 rows
#define BC        (BATCH*NCHUNK)     // 16 (b,c) pairs
#define EPSF      1e-5f

typedef unsigned short u16;
typedef __attribute__((ext_vector_type(16))) __bf16 bf16x16;
typedef __attribute__((ext_vector_type(8)))  float  floatx8;

__device__ __forceinline__ float bf2f(u16 h) {
    unsigned int u = ((unsigned int)h) << 16;
    float f; __builtin_memcpy(&f, &u, 4); return f;
}
__device__ __forceinline__ u16 f2bf(float f) {
    unsigned int u; __builtin_memcpy(&u, &f, 4);
    u = (u + 0x7FFFu + ((u >> 16) & 1u)) >> 16;
    return (u16)u;
}

// A fragment (16x32 bf16, ISA 7.12.2): lane row = lane&15, half = lane>>4.
// elems 0..7  = K[half*8 .. half*8+8), elems 8..15 = K[16+half*8 .. +8).
__device__ __forceinline__ bf16x16 load_frag_a(const u16* p) {
    union { uint4 u[2]; bf16x16 v; } f;
    f.u[0] = *(const uint4*)p;
    f.u[1] = *(const uint4*)(p + 16);
    return f.v;
}
// B fragment (32x16 bf16): lane col = lane&15, half = lane>>4,
// elems 0..7 = K[half*8 .. +8), elems 8..15 = K[16+half*8 .. +8) in [N][K] storage.
__device__ __forceinline__ bf16x16 load_frag_b(const u16* p) {
    union { uint4 u[2]; bf16x16 v; } f;
    f.u[0] = *(const uint4*)p;
    f.u[1] = *(const uint4*)(p + 16);
    return f.v;
}

#define WMMA_BF16(A, B, C) \
    __builtin_amdgcn_wmma_f32_16x16x32_bf16(false, (A), false, (B), (short)0, (C), false, false)

// ---------------------------------------------------------------------------
// K1: f32 -> bf16 cast
// ---------------------------------------------------------------------------
__global__ __launch_bounds__(256) void cast_kernel(const float* __restrict__ in,
                                                   u16* __restrict__ out, long n) {
    long gid = (long)blockIdx.x * 256 + threadIdx.x;
    if (gid < n) out[gid] = f2bf(in[gid]);
}

// ---------------------------------------------------------------------------
// K2: tiled WMMA GEMM  C[M,N] (+batch) = A[M,K]bf16 @ W[N,K]bf16^T
//     BM=128, BN=64, BK=64; 8 waves, each 32x32 (2x2 WMMA tiles).
//     Double-buffered LDS with register staging (1 barrier / 64-deep K step),
//     global_prefetch two stages ahead. OUT_BF16 selects f32 or bf16 output.
// ---------------------------------------------------------------------------
template <int OUT_BF16>
__global__ __launch_bounds__(256) void gemm_kernel(const u16* __restrict__ A,
                                                   const u16* __restrict__ W,
                                                   void* __restrict__ Cout,
                                                   int M, int N, int K,
                                                   long strideA, long strideW, long strideC) {
    // 144B row stride: 16B aligned (ds_load_b128) and conflict-free mod-64 banks.
    __shared__ __attribute__((aligned(16))) u16 sA[2][128][72];
    __shared__ __attribute__((aligned(16))) u16 sB[2][64][72];

    const u16* Ab = A + (size_t)blockIdx.z * strideA;
    const u16* Wb = W + (size_t)blockIdx.z * strideW;

    const int bm = blockIdx.y, bn = blockIdx.x, tid = threadIdx.x;
    const int lane = tid & 31, wid = tid >> 5;
    const int wm = wid >> 1, wn = wid & 1;        // waves: 4(M) x 2(N)
    const int half = lane >> 4, mrow = lane & 15;

    floatx8 acc[2][2] = {};

    const u16* Ap = Ab + (size_t)bm * 128 * K;
    const u16* Wp = Wb + (size_t)bn * 64 * K;
    const int lar = tid >> 1, lac = (tid & 1) * 32;  // A: 128 rows x 64, 32 elems/thr
    const int lbr = tid >> 2, lbc = (tid & 3) * 16;  // B:  64 rows x 64, 16 elems/thr

    uint4 ra0, ra1, ra2, ra3, rb0, rb1;

#define GLOAD(K0)                                                          \
    {                                                                      \
        const u16* pa = Ap + (size_t)lar * K + (K0) + lac;                 \
        ra0 = *(const uint4*)pa;        ra1 = *(const uint4*)(pa + 8);     \
        ra2 = *(const uint4*)(pa + 16); ra3 = *(const uint4*)(pa + 24);    \
        const u16* pb = Wp + (size_t)lbr * K + (K0) + lbc;                 \
        rb0 = *(const uint4*)pb;        rb1 = *(const uint4*)(pb + 8);     \
    }
#define SSTORE(BUF)                                                        \
    {                                                                      \
        *(uint4*)&sA[BUF][lar][lac]      = ra0;                            \
        *(uint4*)&sA[BUF][lar][lac + 8]  = ra1;                            \
        *(uint4*)&sA[BUF][lar][lac + 16] = ra2;                            \
        *(uint4*)&sA[BUF][lar][lac + 24] = ra3;                            \
        *(uint4*)&sB[BUF][lbr][lbc]      = rb0;                            \
        *(uint4*)&sB[BUF][lbr][lbc + 8]  = rb1;                            \
    }

    const int nk = K >> 6;           // number of 64-deep stages
    GLOAD(0);
    SSTORE(0);
    __syncthreads();

    for (int ks = 0; ks < nk; ++ks) {
        const int cur = ks & 1;
        if (ks + 1 < nk) GLOAD((ks + 1) << 6);           // overlap with WMMA below
        if (ks + 2 < nk) {                               // global_prefetch_b8, 2 ahead
            __builtin_prefetch(Ap + (size_t)lar * K + ((ks + 2) << 6) + lac, 0, 1);
            __builtin_prefetch(Wp + (size_t)lbr * K + ((ks + 2) << 6) + lbc, 0, 1);
        }
        for (int kk = 0; kk < 2; ++kk) {                 // 2 x 32-deep WMMA sub-steps
            bf16x16 af[2], bfv[2];
            for (int i = 0; i < 2; ++i)
                af[i] = load_frag_a(&sA[cur][wm * 32 + i * 16 + mrow][kk * 32 + half * 8]);
            for (int j = 0; j < 2; ++j)
                bfv[j] = load_frag_b(&sB[cur][wn * 32 + j * 16 + mrow][kk * 32 + half * 8]);
            for (int i = 0; i < 2; ++i)
                for (int j = 0; j < 2; ++j)
                    acc[i][j] = WMMA_BF16(af[i], bfv[j], acc[i][j]);
        }
        if (ks + 1 < nk) SSTORE(cur ^ 1);   // other buffer: safe w/ single barrier
        __syncthreads();
    }
#undef GLOAD
#undef SSTORE

    const int ch = lane >> 4, cn = lane & 15;
    for (int i = 0; i < 2; ++i)
        for (int j = 0; j < 2; ++j) {
            int row0 = bm * 128 + wm * 32 + i * 16 + ch * 8;
            int col  = bn * 64 + wn * 32 + j * 16 + cn;
            if (OUT_BF16) {
                u16* Cb = (u16*)Cout + (size_t)blockIdx.z * strideC;
                for (int r = 0; r < 8; ++r) Cb[(size_t)(row0 + r) * N + col] = f2bf(acc[i][j][r]);
            } else {
                float* Cf = (float*)Cout + (size_t)blockIdx.z * strideC;
                for (int r = 0; r < 8; ++r) Cf[(size_t)(row0 + r) * N + col] = acc[i][j][r];
            }
        }
}

// ---------------------------------------------------------------------------
// K3: dt = softplus(zx[..., 8448+h] + dt_bias[h])   -> dt[b,l,h] f32
// ---------------------------------------------------------------------------
__global__ __launch_bounds__(256) void dt_kernel(const float* __restrict__ zx,
                                                 const float* __restrict__ dt_bias,
                                                 float* __restrict__ dt) {
    int gid = blockIdx.x * 256 + threadIdx.x;          // BL*NHEADS
    int h = gid & 63;
    long bl = gid >> 6;
    float x = zx[(size_t)bl * D_IN_PROJ + (D_INNER + CONV_DIM) + h] + dt_bias[h];
    dt[gid] = (x > 20.f) ? x : log1pf(expf(x));
}

// ---------------------------------------------------------------------------
// K4: dAcs[b,h,c,l] = cumsum_l( dt * (-exp(A_log[h])) )  (per-chunk scan)
// ---------------------------------------------------------------------------
__global__ __launch_bounds__(256) void dacs_kernel(const float* __restrict__ dt,
                                                   const float* __restrict__ A_log,
                                                   float* __restrict__ dAcs) {
    int blk = blockIdx.x;          // b*512 + h*8 + c
    int c = blk & 7, h = (blk >> 3) & 63, b = blk >> 9;
    int t = threadIdx.x;
    float Ah = -expf(A_log[h]);
    int l = c * CHUNK + t;
    __shared__ float s[256];
    s[t] = dt[((size_t)b * SEQLEN + l) * NHEADS + h] * Ah;
    __syncthreads();
    for (int off = 1; off < 256; off <<= 1) {
        float add = (t >= off) ? s[t - off] : 0.f;
        __syncthreads();
        s[t] += add;
        __syncthreads();
    }
    dAcs[(size_t)blk * 256 + t] = s[t];
}

// ---------------------------------------------------------------------------
// K5: depthwise conv(4) + silu -> xconv[b,l,ch] f32; also bf16 B/C packs
// ---------------------------------------------------------------------------
__global__ __launch_bounds__(256) void conv_kernel(const float* __restrict__ zx,
                                                   const float* __restrict__ conv_w,
                                                   const float* __restrict__ conv_b,
                                                   float* __restrict__ xconv,
                                                   u16* __restrict__ Bb, u16* __restrict__ Cb) {
    long gid = (long)blockIdx.x * 256 + threadIdx.x;   // BATCH*SEQLEN*CONV_DIM
    int ch = (int)(gid % CONV_DIM);
    long bl = gid / CONV_DIM;
    int l = (int)(bl % SEQLEN);
    int b = (int)(bl / SEQLEN);
    float acc = conv_b[ch];
    for (int k = 0; k < D_CONV; ++k) {
        int li = l - (D_CONV - 1) + k;
        if (li >= 0)
            acc += conv_w[ch * D_CONV + k] *
                   zx[((size_t)b * SEQLEN + li) * D_IN_PROJ + D_INNER + ch];
    }
    float s = acc / (1.f + expf(-acc));
    xconv[gid] = s;
    if (ch >= D_INNER) {
        u16 hv = f2bf(s);
        if (ch < D_INNER + D_STATE) Bb[(size_t)bl * D_STATE + (ch - D_INNER)] = hv;
        else                        Cb[(size_t)bl * D_STATE + (ch - D_INNER - D_STATE)] = hv;
    }
}

// ---------------------------------------------------------------------------
// K6: B^T pack:  BT[bc,n,s] = Bb[bc,s,n]   ([N][K] layout for WMMA B-operand)
// ---------------------------------------------------------------------------
__global__ __launch_bounds__(256) void packbt_kernel(const u16* __restrict__ Bb,
                                                     u16* __restrict__ BT) {
    int gid = blockIdx.x * 256 + threadIdx.x;          // BC*128*256
    int s = gid & 255, n = (gid >> 8) & 127, bc = gid >> 15;
    BT[gid] = Bb[((size_t)bc * CHUNK + s) * D_STATE + n];
}

// ---------------------------------------------------------------------------
// K7: xdt packs (transposed [p][s] layouts):
//     xdtT[bc,h,p,s]    = x[b,l,h,p]*dt
//     xdtdecT[bc,h,p,s] = x*dt*exp(dAcs[255]-dAcs[s])
// ---------------------------------------------------------------------------
__global__ __launch_bounds__(256) void packxdt_kernel(const float* __restrict__ xconv,
                                                      const float* __restrict__ dt,
                                                      const float* __restrict__ dAcs,
                                                      u16* __restrict__ xdtT,
                                                      u16* __restrict__ xdtdecT) {
    int blk = blockIdx.x;                               // bc*64 + h
    int h = blk & 63, bc = blk >> 6, c = bc & 7, b = bc >> 3;
    const float* da = dAcs + ((size_t)(b * 64 + h) * 8 + c) * 256;
    float daL = da[255];
    int t = threadIdx.x;
    int p = t >> 2, s0 = (t & 3) * 64;
    size_t base = ((size_t)blk * HEADDIM + p) * CHUNK;
    for (int i = 0; i < 64; ++i) {
        int s = s0 + i;
        int l = c * CHUNK + s;
        float x = xconv[((size_t)b * SEQLEN + l) * CONV_DIM + h * HEADDIM + p];
        float d = dt[((size_t)b * SEQLEN + l) * NHEADS + h];
        float xd = x * d;
        xdtT[base + s]    = f2bf(xd);
        xdtdecT[base + s] = f2bf(xd * expf(daL - da[s]));
    }
}

// ---------------------------------------------------------------------------
// K9: states[bc,h,p,n] = sum_s xdtdecT[p,s] * B[s,n]   (M=64,N=128,K=256)
//     direct-from-global WMMA fragments (operands are small, live in L2)
// ---------------------------------------------------------------------------
__global__ __launch_bounds__(256) void states_kernel(const u16* __restrict__ xdtdecT,
                                                     const u16* __restrict__ BT,
                                                     float* __restrict__ states) {
    int blk = blockIdx.x;                               // bc*64 + h
    int bc = blk >> 6;
    const u16* Abase = xdtdecT + (size_t)blk * HEADDIM * CHUNK;   // [p][s]
    const u16* Bbase = BT + (size_t)bc * D_STATE * CHUNK;         // [n][s]
    int tid = threadIdx.x, lane = tid & 31, wid = tid >> 5;
    int wm = wid >> 1, wn = wid & 1;                    // 4(M) x 2(N) waves, wave: 16x64
    int half = lane >> 4, mrow = lane & 15;
    floatx8 acc[4] = {};
    for (int k0 = 0; k0 < CHUNK; k0 += 32) {
        bf16x16 af = load_frag_a(Abase + (size_t)(wm * 16 + mrow) * CHUNK + k0 + half * 8);
        for (int j = 0; j < 4; ++j) {
            int ncol = (wn * 4 + j) * 16 + mrow;
            bf16x16 bfv = load_frag_b(Bbase + (size_t)ncol * CHUNK + k0 + half * 8);
            acc[j] = WMMA_BF16(af, bfv, acc[j]);
        }
    }
    float* out = states + (size_t)blk * HEADDIM * D_STATE;
    int row0 = wm * 16 + (lane >> 4) * 8, cn = lane & 15;
    for (int j = 0; j < 4; ++j) {
        int col = (wn * 4 + j) * 16 + cn;
        for (int r = 0; r < 8; ++r) out[(size_t)(row0 + r) * D_STATE + col] = acc[j][r];
    }
}

// ---------------------------------------------------------------------------
// K10: inter-chunk scan: prev[c] = carry; carry = carry*dec[c] + states[c]
// ---------------------------------------------------------------------------
__global__ __launch_bounds__(256) void scan_kernel(const float* __restrict__ states,
                                                   const float* __restrict__ dAcs,
                                                   u16* __restrict__ prevb) {
    int blk = blockIdx.x;                               // b*64 + h
    int b = blk >> 6, h = blk & 63;
    float dec[NCHUNK];
    for (int c = 0; c < NCHUNK; ++c)
        dec[c] = expf(dAcs[((size_t)(b * 64 + h) * 8 + c) * 256 + 255]);
    int t = threadIdx.x;
    for (int i = 0; i < 32; ++i) {                      // 8192 (p,n) elems / 256 thr
        int pn = i * 256 + t;
        float carry = 0.f;
        for (int c = 0; c < NCHUNK; ++c) {
            size_t idx = ((size_t)((b * 8 + c) * 64 + h)) * (HEADDIM * D_STATE) + pn;
            prevb[idx] = f2bf(carry);
            carry = carry * dec[c] + states[idx];
        }
    }
}

// ---------------------------------------------------------------------------
// K11: per (b,c,h,ltile): Y[l,p] = (CB∘L)@xdt + (C·e^{dAcs})@prev + D_skip*x
//      M1/A2 built in LDS, WMMA from LDS(A) x global(B).
// ---------------------------------------------------------------------------
__global__ __launch_bounds__(256) void y_kernel(const u16* __restrict__ CBb,
                                                const u16* __restrict__ Cb,
                                                const u16* __restrict__ xdtT,
                                                const u16* __restrict__ prevb,
                                                const float* __restrict__ dAcs,
                                                const float* __restrict__ xconv,
                                                const float* __restrict__ D_skip,
                                                float* __restrict__ Y) {
    int blk = blockIdx.x;                               // lt(2b) | h(6b) | bc(4b)
    int lt4 = blk & 3, h = (blk >> 2) & 63, bc = blk >> 8, b = bc >> 3, c = bc & 7;

    __shared__ __attribute__((aligned(16))) u16 sM1[64][264];   // masked decay * CB
    __shared__ __attribute__((aligned(16))) u16 sA2[64][136];   // C * exp(dAcs[l])
    __shared__ float sDA[256];

    int t = threadIdx.x;
    sDA[t] = dAcs[((size_t)(b * 64 + h) * 8 + c) * 256 + t];
    __syncthreads();

    {
        int lt = t >> 2;
        int l = lt4 * 64 + lt;
        float dal = sDA[l];
        const u16* cbrow = CBb + ((size_t)bc * CHUNK + l) * CHUNK;
        int s0 = (t & 3) * 64;
        for (int i = 0; i < 64; ++i) {
            int s = s0 + i;
            float v = (s <= l) ? expf(dal - sDA[s]) * bf2f(cbrow[s]) : 0.f;
            sM1[lt][s] = f2bf(v);
        }
        float el = expf(dal);
        const u16* crow = Cb + ((size_t)bc * CHUNK + l) * D_STATE;
        int n0 = (t & 3) * 32;
        for (int i = 0; i < 32; ++i) {
            int n = n0 + i;
            sA2[lt][n] = f2bf(bf2f(crow[n]) * el);
        }
    }
    __syncthreads();

    int lane = t & 31, wid = t >> 5;
    int wm = wid >> 1, wn = wid & 1;                    // waves 4(M) x 2(N), wave: 16x32
    int half = lane >> 4, mrow = lane & 15;
    floatx8 acc[2] = {};

    const u16* xb = xdtT + (size_t)(bc * 64 + h) * HEADDIM * CHUNK;   // [p][s]
    for (int k0 = 0; k0 < CHUNK; k0 += 32) {
        bf16x16 af = load_frag_a(&sM1[wm * 16 + mrow][k0 + half * 8]);
        for (int j = 0; j < 2; ++j) {
            int pcol = (wn * 2 + j) * 16 + mrow;
            bf16x16 bfv = load_frag_b(xb + (size_t)pcol * CHUNK + k0 + half * 8);
            acc[j] = WMMA_BF16(af, bfv, acc[j]);
        }
    }
    const u16* pb = prevb + (size_t)(bc * 64 + h) * HEADDIM * D_STATE;   // [p][n]
    for (int k0 = 0; k0 < D_STATE; k0 += 32) {
        bf16x16 af = load_frag_a(&sA2[wm * 16 + mrow][k0 + half * 8]);
        for (int j = 0; j < 2; ++j) {
            int pcol = (wn * 2 + j) * 16 + mrow;
            bf16x16 bfv = load_frag_b(pb + (size_t)pcol * D_STATE + k0 + half * 8);
            acc[j] = WMMA_BF16(af, bfv, acc[j]);
        }
    }

    float dsk = D_skip[h];
    int cn = lane & 15, ch8 = lane >> 4;
    for (int j = 0; j < 2; ++j) {
        int pcol = (wn * 2 + j) * 16 + cn;
        int lrow0 = lt4 * 64 + wm * 16 + ch8 * 8;
        for (int r = 0; r < 8; ++r) {
            int gl = c * CHUNK + lrow0 + r;
            size_t xi = ((size_t)b * SEQLEN + gl) * CONV_DIM + h * HEADDIM + pcol;
            size_t yi = ((size_t)b * SEQLEN + gl) * D_INNER + h * HEADDIM + pcol;
            Y[yi] = acc[j][r] + dsk * xconv[xi];
        }
    }
}

// ---------------------------------------------------------------------------
// K12: y = y * silu(z); RMSNorm over 4096; * norm_w; -> bf16
// ---------------------------------------------------------------------------
__global__ __launch_bounds__(256) void gate_norm_kernel(const float* __restrict__ Y,
                                                        const float* __restrict__ zx,
                                                        const float* __restrict__ norm_w,
                                                        u16* __restrict__ yb) {
    int row = blockIdx.x;                               // b*SEQLEN + l
    int t = threadIdx.x;
    const float* yr = Y + (size_t)row * D_INNER;
    const float* zr = zx + (size_t)row * D_IN_PROJ;
    float v[16], ss = 0.f;
    for (int i = 0; i < 16; ++i) {
        int e = t + i * 256;
        float z = zr[e];
        float val = yr[e] * (z / (1.f + expf(-z)));
        v[i] = val;
        ss += val * val;
    }
    __shared__ float red[256];
    red[t] = ss;
    __syncthreads();
    for (int off = 128; off > 0; off >>= 1) {
        if (t < off) red[t] += red[t + off];
        __syncthreads();
    }
    float scale = rsqrtf(red[0] / (float)D_INNER + EPSF);
    u16* ob = yb + (size_t)row * D_INNER;
    for (int i = 0; i < 16; ++i) {
        int e = t + i * 256;
        ob[e] = f2bf(v[i] * scale * norm_w[e]);
    }
}

// ---------------------------------------------------------------------------
extern "C" void kernel_launch(void* const* d_in, const int* in_sizes, int n_in,
                              void* d_out, int out_size, void* d_ws, size_t ws_size,
                              hipStream_t stream) {
    (void)in_sizes; (void)n_in; (void)out_size; (void)ws_size;
    const float* u         = (const float*)d_in[0];
    const float* in_proj_w = (const float*)d_in[1];
    const float* conv_w    = (const float*)d_in[2];
    const float* conv_b    = (const float*)d_in[3];
    const float* dt_bias   = (const float*)d_in[4];
    const float* A_log     = (const float*)d_in[5];
    const float* D_skip    = (const float*)d_in[6];
    const float* norm_w    = (const float*)d_in[7];
    const float* out_proj_w= (const float*)d_in[8];
    float* out = (float*)d_out;

    char* ws = (char*)d_ws;
    size_t off = 0;
    auto alloc = [&](size_t bytes) { size_t o = off; off += (bytes + 255) & ~(size_t)255; return o; };

    u16*   u_b     = (u16*)  (ws + alloc((size_t)BL * D_MODEL * 2));
    u16*   wi_b    = (u16*)  (ws + alloc((size_t)D_IN_PROJ * D_MODEL * 2));
    u16*   wo_b    = (u16*)  (ws + alloc((size_t)D_MODEL * D_INNER * 2));
    float* zx      = (float*)(ws + alloc((size_t)BL * D_IN_PROJ * 4));
    float* dtp     = (float*)(ws + alloc((size_t)BL * NHEADS * 4));
    float* dAcs    = (float*)(ws + alloc((size_t)BATCH * NHEADS * NCHUNK * CHUNK * 4));
    float* xconv   = (float*)(ws + alloc((size_t)BL * CONV_DIM * 4));
    u16*   Bb      = (u16*)  (ws + alloc((size_t)BL * D_STATE * 2));
    u16*   Cbm     = (u16*)  (ws + alloc((size_t)BL * D_STATE * 2));
    u16*   BT      = (u16*)  (ws + alloc((size_t)BC * D_STATE * CHUNK * 2));
    u16*   xdtT    = (u16*)  (ws + alloc((size_t)BC * NHEADS * HEADDIM * CHUNK * 2));
    u16*   xdtdecT = (u16*)  (ws + alloc((size_t)BC * NHEADS * HEADDIM * CHUNK * 2));
    u16*   CBb     = (u16*)  (ws + alloc((size_t)BC * CHUNK * CHUNK * 2));
    float* states  = (float*)(ws + alloc((size_t)BC * NHEADS * HEADDIM * D_STATE * 4));
    u16*   prevb   = (u16*)  (ws + alloc((size_t)BC * NHEADS * HEADDIM * D_STATE * 2));
    float* Y       = (float*)(ws + alloc((size_t)BL * D_INNER * 4));
    u16*   yb      = (u16*)  (ws + alloc((size_t)BL * D_INNER * 2));

    // casts
    { long n = (long)BL * D_MODEL;
      cast_kernel<<<(n + 255) / 256, 256, 0, stream>>>(u, u_b, n); }
    { long n = (long)D_IN_PROJ * D_MODEL;
      cast_kernel<<<(n + 255) / 256, 256, 0, stream>>>(in_proj_w, wi_b, n); }
    { long n = (long)D_MODEL * D_INNER;
      cast_kernel<<<(n + 255) / 256, 256, 0, stream>>>(out_proj_w, wo_b, n); }

    // in_proj: zx[BL, 8512] = u_b @ wi_b^T
    gemm_kernel<0><<<dim3(D_IN_PROJ / 64, BL / 128, 1), 256, 0, stream>>>(
        u_b, wi_b, zx, BL, D_IN_PROJ, D_MODEL, 0, 0, 0);

    // dt softplus
    dt_kernel<<<(BL * NHEADS) / 256, 256, 0, stream>>>(zx, dt_bias, dtp);
    // per-chunk cumsum of dA
    dacs_kernel<<<BATCH * NHEADS * NCHUNK, 256, 0, stream>>>(dtp, A_log, dAcs);
    // conv + silu (+ B/C bf16 packs)
    conv_kernel<<<((long)BL * CONV_DIM) / 256, 256, 0, stream>>>(zx, conv_w, conv_b, xconv, Bb, Cbm);
    // B^T pack
    packbt_kernel<<<(BC * D_STATE * CHUNK) / 256, 256, 0, stream>>>(Bb, BT);
    // xdt packs
    packxdt_kernel<<<BC * NHEADS, 256, 0, stream>>>(xconv, dtp, dAcs, xdtT, xdtdecT);

    // CB[bc,l,s] = C @ B^T  (batched, bf16 out): M=256,N=256,K=128
    gemm_kernel<1><<<dim3(CHUNK / 64, CHUNK / 128, BC), 256, 0, stream>>>(
        Cbm, Bb, CBb, CHUNK, CHUNK, D_STATE,
        (long)CHUNK * D_STATE, (long)CHUNK * D_STATE, (long)CHUNK * CHUNK);

    // per-chunk states
    states_kernel<<<BC * NHEADS, 256, 0, stream>>>(xdtdecT, BT, states);
    // inter-chunk scan
    scan_kernel<<<BATCH * NHEADS, 256, 0, stream>>>(states, dAcs, prevb);
    // Y = Y_diag + Y_off + skip
    y_kernel<<<BC * NHEADS * 4, 256, 0, stream>>>(CBb, Cbm, xdtT, prevb, dAcs, xconv, D_skip, Y);
    // gate + RMSNorm -> bf16
    gate_norm_kernel<<<BL, 256, 0, stream>>>(Y, zx, norm_w, yb);

    // out_proj: out[BL, 2048] = yb @ wo_b^T
    gemm_kernel<0><<<dim3(D_MODEL / 64, BL / 128, 1), 256, 0, stream>>>(
        yb, wo_b, out, BL, D_MODEL, D_INNER, 0, 0, 0);
}
